// MultiHeadAttention_29343216566721
// MI455X (gfx1250) — matrix operand order repaired and checked by reading, not compiled, not run on hardware
//
#include <hip/hip_runtime.h>
#include <hip/hip_bf16.h>

// Problem constants (B=2, S=2048, D=1024, H=16, DK=64)
#define SB 2
#define SS 2048
#define SD 1024
#define SH 16
#define SDK 64

typedef __attribute__((ext_vector_type(16))) _Float16 v16h;
typedef __attribute__((ext_vector_type(8)))  _Float16 v8h;
typedef __attribute__((ext_vector_type(8)))  float    v8f;

__device__ __forceinline__ v16h cat16(v8h lo, v8h hi) {
    return __builtin_shufflevector(lo, hi, 0,1,2,3,4,5,6,7,8,9,10,11,12,13,14,15);
}

__device__ __forceinline__ v8f wmma16(v16h a, v16h b, v8f c) {
    // D = A(16x32 f16) * B(32x16 f16) + C(16x16 f32)
    return __builtin_amdgcn_wmma_f32_16x16x32_f16(false, a, false, b, (short)0, c, false, false);
}

// ---- fragment loaders ---------------------------------------------------
// A-fragment (16x32, M x K) from row-major fp32 [.., ld], converting to f16.
// lane: row = m0 + (lane&15); K = kc + (lane>>4)*8 + {0..7, 16..23}
__device__ __forceinline__ v16h load_a_f32(const float* __restrict__ X, int ld,
                                           int m0, int kc, int lane) {
    int row = m0 + (lane & 15);
    int k0  = kc + ((lane >> 4) << 3);
    const float* p = X + (size_t)row * ld + k0;
    float4 a0 = *(const float4*)(p);
    float4 a1 = *(const float4*)(p + 4);
    float4 b0 = *(const float4*)(p + 16);
    float4 b1 = *(const float4*)(p + 20);
    v16h r;
    r[0]=(_Float16)a0.x; r[1]=(_Float16)a0.y; r[2]=(_Float16)a0.z; r[3]=(_Float16)a0.w;
    r[4]=(_Float16)a1.x; r[5]=(_Float16)a1.y; r[6]=(_Float16)a1.z; r[7]=(_Float16)a1.w;
    r[8]=(_Float16)b0.x; r[9]=(_Float16)b0.y; r[10]=(_Float16)b0.z; r[11]=(_Float16)b0.w;
    r[12]=(_Float16)b1.x; r[13]=(_Float16)b1.y; r[14]=(_Float16)b1.z; r[15]=(_Float16)b1.w;
    return r;
}

// A-fragment from row-major f16 [.., ld]
__device__ __forceinline__ v16h load_a_f16(const _Float16* __restrict__ X, int ld,
                                           int m0, int kc, int lane) {
    int row = m0 + (lane & 15);
    int k0  = kc + ((lane >> 4) << 3);
    const _Float16* p = X + (size_t)row * ld + k0;
    v8h lo = *(const v8h*)(p);
    v8h hi = *(const v8h*)(p + 16);
    return cat16(lo, hi);
}

// B-fragment (32x16, K x N) where B^T is row-major [N rows, K cols] fp32
// (i.e. W[n][k] for Y = X*W^T). lane: n = n0+(lane&15); K = kc + (lane>>4)*8 + {0..7, 16..23}
__device__ __forceinline__ v16h load_b_f32(const float* __restrict__ W, int ld,
                                           int n0, int kc, int lane) {
    int n  = n0 + (lane & 15);
    int k0 = kc + ((lane >> 4) << 3);
    const float* p = W + (size_t)n * ld + k0;
    float4 x0 = *(const float4*)(p);
    float4 x1 = *(const float4*)(p + 4);
    float4 x2 = *(const float4*)(p + 16);
    float4 x3 = *(const float4*)(p + 20);
    v16h r;
    r[0]=(_Float16)x0.x; r[1]=(_Float16)x0.y; r[2]=(_Float16)x0.z; r[3]=(_Float16)x0.w;
    r[4]=(_Float16)x1.x; r[5]=(_Float16)x1.y; r[6]=(_Float16)x1.z; r[7]=(_Float16)x1.w;
    r[8]=(_Float16)x2.x; r[9]=(_Float16)x2.y; r[10]=(_Float16)x2.z; r[11]=(_Float16)x2.w;
    r[12]=(_Float16)x3.x; r[13]=(_Float16)x3.y; r[14]=(_Float16)x3.z; r[15]=(_Float16)x3.w;
    return r;
}

// B-fragment where B^T is row-major f16 [N rows, K cols]  (K matrix [S,DK] for QK^T,
// or V^T [DK,S] for PV).
__device__ __forceinline__ v16h load_b_f16(const _Float16* __restrict__ M, int ld,
                                           int n0, int kc, int lane) {
    int n  = n0 + (lane & 15);
    int k0 = kc + ((lane >> 4) << 3);
    const _Float16* p = M + (size_t)n * ld + k0;
    v8h lo = *(const v8h*)(p);
    v8h hi = *(const v8h*)(p + 16);
    return cat16(lo, hi);
}

// ---- kernel 1: projection  Y = X @ W^T + b, f32 in -> f16 out -----------
// Each wave computes a 16x64 output strip: one A-fragment reused across 4
// B-fragments / 4 accumulators (4x less A traffic + f32->f16 convert VALU).
// TRANS==0: Y layout [B,H,S,DK].  TRANS==1: Y layout [B,H,DK,S].
template<int TRANS>
__global__ __launch_bounds__(128)
void proj_kernel(const float* __restrict__ X, const float* __restrict__ W,
                 const float* __restrict__ bias, _Float16* __restrict__ Y) {
    int lane = threadIdx.x & 31;
    int w    = blockIdx.x * 4 + (threadIdx.x >> 5);   // 4096 waves
    int mt   = w >> 4;            // M/16  = 256
    int ng   = w & 15;            // N/64  = 16
    int m0  = mt << 4;
    int n0g = ng << 6;

    v8f acc[4] = {{}, {}, {}, {}};
    for (int kc = 0; kc < SD; kc += 32) {
        v16h a = load_a_f32(X, SD, m0, kc, lane);
#pragma unroll
        for (int j = 0; j < 4; ++j)
            acc[j] = wmma16(a, load_b_f32(W, SD, n0g + (j << 4), kc, lane), acc[j]);
    }

    int rbase = m0 + ((lane >> 4) << 3);
#pragma unroll
    for (int j = 0; j < 4; ++j) {
        int n  = n0g + (j << 4) + (lane & 15);
        float bb = bias[n];
        int h  = n >> 6;
        int dk = n & (SDK - 1);
#pragma unroll
        for (int e = 0; e < 8; ++e) {
            int m  = rbase + e;
            int bi = m >> 11;           // m / S
            int s  = m & (SS - 1);
            size_t idx;
            if (TRANS)
                idx = (((size_t)bi * SH + h) * SDK + dk) * SS + s;
            else
                idx = (((size_t)bi * SH + h) * SS + s) * SDK + dk;
            Y[idx] = (_Float16)(acc[j][e] + bb);
        }
    }
}

// ---- kernel 2: fused causal attention -----------------------------------
// One wave per (b,h, 16-query block).  Pass1: online (max,sumexp) partials.
// Pass2: recompute scores, write normalized attn (f32), transpose P via LDS,
// accumulate O = P @ V with WMMA against V^T.
__global__ __launch_bounds__(128)
void attn_kernel(const _Float16* __restrict__ Q, const _Float16* __restrict__ K,
                 const _Float16* __restrict__ VT, float* __restrict__ attn_out,
                 _Float16* __restrict__ O) {
    __shared__ float lds[4][16 * 32];    // per-wave 16x32 P tile (8 KB total)
    int lane = threadIdx.x & 31;
    int wv   = threadIdx.x >> 5;
    int w    = blockIdx.x * 4 + wv;      // 4096 waves total
    int qt = w & 127;                    // query tile (S/16 = 128)
    int h  = (w >> 7) & (SH - 1);
    int b  = w >> 11;

    const _Float16* Qh = Q  + (((size_t)b * SH + h) * SS) * SDK;
    const _Float16* Kh = K  + (((size_t)b * SH + h) * SS) * SDK;
    const _Float16* Vh = VT + (((size_t)b * SH + h) * SDK) * SS;
    int qbase = qt << 4;
    float* A_out = attn_out + ((size_t)(b * SH + h)) * SS * SS + (size_t)qbase * SS;

    v16h aq0 = load_a_f16(Qh, SDK, qbase, 0,  lane);
    v16h aq1 = load_a_f16(Qh, SDK, qbase, 32, lane);

    int rloc0 = (lane >> 4) << 3;        // local row of accumulator element 0
    int col_t = lane & 15;               // local column within a key tile
    const float scale = 0.125f;          // 1/sqrt(64)

    float m[8], lsum[8];
#pragma unroll
    for (int e = 0; e < 8; ++e) { m[e] = -3.0e38f; lsum[e] = 0.0f; }

    // ---- pass 1: per-lane online softmax stats over valid key tiles ----
    for (int t = 0; t <= qt; ++t) {
        if (t < qt)  // pull next K tile rows toward the WGP cache
            __builtin_prefetch(Kh + (size_t)(((t + 1) << 4) + (lane & 15)) * SDK, 0, 3);
        v8f c = {};
        c = wmma16(aq0, load_b_f16(Kh, SDK, t << 4, 0,  lane), c);
        c = wmma16(aq1, load_b_f16(Kh, SDK, t << 4, 32, lane), c);
        bool diag = (t == qt);
#pragma unroll
        for (int e = 0; e < 8; ++e) {
            float s = c[e] * scale;
            if (diag && ((t << 4) + col_t > qbase + rloc0 + e)) s = -1e9f;
            float mo = m[e];
            float mn = fmaxf(mo, s);
            lsum[e] = lsum[e] * __expf(mo - mn) + __expf(s - mn);
            m[e] = mn;
        }
    }
    // merge partials across the 16 lanes that share each row
#pragma unroll
    for (int e = 0; e < 8; ++e) {
#pragma unroll
        for (int msk = 1; msk < 16; msk <<= 1) {
            float mo = __shfl_xor(m[e],    msk, 16);
            float lo = __shfl_xor(lsum[e], msk, 16);
            float mn = fmaxf(m[e], mo);
            lsum[e] = lsum[e] * __expf(m[e] - mn) + lo * __expf(mo - mn);
            m[e] = mn;
        }
    }
    float invl[8];
#pragma unroll
    for (int e = 0; e < 8; ++e) invl[e] = 1.0f / lsum[e];

    // ---- pass 2: normalized attn out + O = P @ V ----
    float* lp = lds[wv];
    v8f acc[4] = {{}, {}, {}, {}};
    int nchunks = (qt >> 1) + 1;         // 32-key chunks containing valid tiles
    for (int tt = 0; tt < nchunks; ++tt) {
#pragma unroll
        for (int half = 0; half < 2; ++half) {
            int t = (tt << 1) + half;
            if (t <= qt) {
                v8f c = {};
                c = wmma16(aq0, load_b_f16(Kh, SDK, t << 4, 0,  lane), c);
                c = wmma16(aq1, load_b_f16(Kh, SDK, t << 4, 32, lane), c);
                bool diag = (t == qt);
#pragma unroll
                for (int e = 0; e < 8; ++e) {
                    int row = rloc0 + e;
                    int col = (t << 4) + col_t;
                    float p;
                    if (diag && col > qbase + row) p = 0.0f;
                    else p = __expf(c[e] * scale - m[e]) * invl[e];
                    A_out[(size_t)row * SS + col] = p;
                    lp[row * 32 + (half << 4) + col_t] = p;
                }
            } else {
#pragma unroll
                for (int e = 0; e < 8; ++e)
                    lp[(rloc0 + e) * 32 + (half << 4) + col_t] = 0.0f;
            }
        }
        asm volatile("s_wait_dscnt 0" ::: "memory");   // LDS P tile visible wave-wide
        // re-load P in A-fragment layout, f32 -> f16
        int prow = lane & 15;
        int pk0  = (lane >> 4) << 3;
        v16h ap;
#pragma unroll
        for (int i = 0; i < 8; ++i) ap[i]     = (_Float16)lp[prow * 32 + pk0 + i];
#pragma unroll
        for (int i = 0; i < 8; ++i) ap[8 + i] = (_Float16)lp[prow * 32 + 16 + pk0 + i];
#pragma unroll
        for (int j = 0; j < 4; ++j)
            acc[j] = wmma16(ap, load_b_f16(Vh, SS, j << 4, tt << 5, lane), acc[j]);
    }

    // zero-fill fully-masked attn columns (vectorized b128 stores)
    int c0 = (qt + 1) << 4;
    if (c0 < SS) {
        int qpr = (SS - c0) >> 2;        // float4s per row
        int total = 16 * qpr;
        float4 z = make_float4(0.f, 0.f, 0.f, 0.f);
        for (int idx = lane; idx < total; idx += 32) {
            int r  = idx / qpr;
            int q4 = idx - r * qpr;
            *(float4*)(&A_out[(size_t)r * SS + c0 + (q4 << 2)]) = z;
        }
    }

    // store O tile (head-merged [B,S,D] f16)
#pragma unroll
    for (int j = 0; j < 4; ++j) {
#pragma unroll
        for (int e = 0; e < 8; ++e) {
            int qq = qbase + rloc0 + e;
            int d  = h * SDK + (j << 4) + (lane & 15);
            O[((size_t)b * SS + qq) * SD + d] = (_Float16)acc[j][e];
        }
    }
}

// ---- kernel 3: output projection  x = O @ Wo^T + bo  (f16 in, f32 out) --
// Same 16x64-per-wave register blocking as proj_kernel.
__global__ __launch_bounds__(128)
void outproj_kernel(const _Float16* __restrict__ O, const float* __restrict__ Wo,
                    const float* __restrict__ bo, float* __restrict__ Xout) {
    int lane = threadIdx.x & 31;
    int w    = blockIdx.x * 4 + (threadIdx.x >> 5);   // 4096 waves
    int mt = w >> 4, ng = w & 15;
    int m0 = mt << 4, n0g = ng << 6;

    v8f acc[4] = {{}, {}, {}, {}};
    for (int kc = 0; kc < SD; kc += 32) {
        v16h a = load_a_f16(O, SD, m0, kc, lane);
#pragma unroll
        for (int j = 0; j < 4; ++j)
            acc[j] = wmma16(a, load_b_f32(Wo, SD, n0g + (j << 4), kc, lane), acc[j]);
    }
    int rbase = m0 + ((lane >> 4) << 3);
#pragma unroll
    for (int j = 0; j < 4; ++j) {
        int n = n0g + (j << 4) + (lane & 15);
        float bb = bo[n];
#pragma unroll
        for (int e = 0; e < 8; ++e)
            Xout[(size_t)(rbase + e) * SD + n] = acc[j][e] + bb;
    }
}

extern "C" void kernel_launch(void* const* d_in, const int* in_sizes, int n_in,
                              void* d_out, int out_size, void* d_ws, size_t ws_size,
                              hipStream_t stream) {
    const float* q  = (const float*)d_in[0];
    const float* k  = (const float*)d_in[1];
    const float* v  = (const float*)d_in[2];
    // d_in[3] = causal mask (known analytically, unused)
    const float* wq = (const float*)d_in[4];
    const float* bq = (const float*)d_in[5];
    const float* wk = (const float*)d_in[6];
    const float* bk = (const float*)d_in[7];
    const float* wv = (const float*)d_in[8];
    const float* bv = (const float*)d_in[9];
    const float* wo = (const float*)d_in[10];
    const float* bo = (const float*)d_in[11];

    const size_t BSD = (size_t)SB * SS * SD;   // 4,194,304
    _Float16* Qw = (_Float16*)d_ws;
    _Float16* Kw = Qw + BSD;
    _Float16* Vt = Kw + BSD;
    _Float16* Ow = Vt + BSD;                   // total 32 MB f16 workspace

    float* xout = (float*)d_out;
    float* attn = xout + BSD;

    dim3 blk(128);
    proj_kernel<0><<<1024, blk, 0, stream>>>(q, wq, bq, Qw);
    proj_kernel<0><<<1024, blk, 0, stream>>>(k, wk, bk, Kw);
    proj_kernel<1><<<1024, blk, 0, stream>>>(v, wv, bv, Vt);
    attn_kernel<<<1024, blk, 0, stream>>>(Qw, Kw, Vt, attn, Ow);
    outproj_kernel<<<1024, blk, 0, stream>>>(Ow, wo, bo, xout);
}
